// LSTMModel_23184233464617
// MI455X (gfx1250) — matrix operand order repaired and checked by reading, not compiled, not run on hardware
//
#include <hip/hip_runtime.h>
#include <hip/hip_bf16.h>

// Problem constants (from reference): B=512, T=2048, D=64, H=66
#define BATCH 512
#define SEQT  2048
#define DIM   64
#define HID   66
#define G4    264              // 4*H gate columns
#define NP    272              // gate columns padded to 17 WMMA N-tiles
#define NT    17               // N tiles (one per wave)
#define KT    5                // K tiles of 32: x -> K 0..63, h -> K 64..129, pad ..159
#define AST   168              // LDS row stride (bf16 elems) for A operand, padded
#define ROWS  16               // batch rows per workgroup (WMMA M)
#define THREADS (NT*32)        // 544 threads = 17 wave32

typedef __attribute__((ext_vector_type(16))) __bf16 v16bf;
typedef __attribute__((ext_vector_type(8)))  float  v8f;

union Frag16 {
  unsigned int u[8];
  uint4 q[2];
  v16bf v;
};

__device__ __forceinline__ unsigned short f2bf(float f) {
  unsigned int u = __float_as_uint(f);
  u += 0x7FFFu + ((u >> 16) & 1u);   // round-to-nearest-even
  return (unsigned short)(u >> 16);
}
// Branch-free transcendentals: v_exp_f32 + v_rcp_f32, no exec-mask churn.
__device__ __forceinline__ float fsig(float x) {
  return __builtin_amdgcn_rcpf(1.0f + __expf(-x));
}
__device__ __forceinline__ float ftanh(float x) {
  // 1 - 2/(e^{2x}+1): exact limits (+1 at +inf via e=inf, -1 at -inf via e=0)
  return 1.0f - 2.0f * __builtin_amdgcn_rcpf(__expf(2.0f * x) + 1.0f);
}

// Fused (input-projection + recurrence) LSTM. One WG owns 16 batch rows and
// iterates all T steps with weights resident in VGPR B-fragments.
__global__ void __launch_bounds__(THREADS, 1)
lstm_fused(const float* __restrict__ x,   const float* __restrict__ Wih,
           const float* __restrict__ Whh, const float* __restrict__ bih,
           const float* __restrict__ bhh, float* __restrict__ hOut) {
  __shared__ __align__(16) unsigned short Abuf[2][ROWS][AST]; // bf16 [x_t | h_t | pad]
  __shared__ float gatesL[ROWS][NP];

  const int tid  = threadIdx.x;
  const int wid  = tid >> 5;
  const int lane = tid & 31;
  const int b0   = blockIdx.x * ROWS;

  // ---- build per-wave B-matrix (weight) fragments: column-major W^T tiles ----
  // B k-order used here (contiguous, A matches): N = lane&15; lanes 0-15 hold K 0..15 (2/VGPR), lanes 16-31 K 16..31
  const int nl = lane & 15;
  const int ng = wid * 16 + nl;        // global gate column
  const int kb = (lane >> 4) * 16;     // K sub-block within a 32-deep k-tile
  Frag16 bf[KT];
  #pragma unroll
  for (int kt = 0; kt < KT; ++kt) {
    #pragma unroll
    for (int j = 0; j < 8; ++j) {
      int k0 = kt*32 + kb + 2*j;
      float w0 = 0.f, w1 = 0.f;
      if (ng < G4) {
        if (k0 < DIM)          w0 = Wih[ng*DIM + k0];
        else if (k0 < DIM+HID) w0 = Whh[ng*HID + (k0-DIM)];
        int k1 = k0 + 1;
        if (k1 < DIM)          w1 = Wih[ng*DIM + k1];
        else if (k1 < DIM+HID) w1 = Whh[ng*HID + (k1-DIM)];
      }
      bf[kt].u[j] = (unsigned)f2bf(w0) | ((unsigned)f2bf(w1) << 16);
    }
  }
  const float bsum = (ng < G4) ? (bih[ng] + bhh[ng]) : 0.0f;   // bias folded into C init

  // ---- hoisted per-thread index math (static across all 2048 steps) ----
  // x staging: elements i = tid, tid+THREADS of the 16x64 x-tile
  bool xv1 = (tid + THREADS) < ROWS*DIM;          // slot 1 valid (tid < 480)
  int  xr0 = tid >> 6,            xk0 = tid & (DIM-1);
  int  xr1 = (tid+THREADS) >> 6,  xk1 = (tid+THREADS) & (DIM-1);
  const float* xp0 = &x[((size_t)(b0+xr0)*SEQT + 1)*DIM + xk0];  // time t=1
  const float* xp1 = &x[((size_t)(b0+xr1)*SEQT + 1)*DIM + xk1];
  // elementwise: cells i = tid, tid+THREADS of the 16x66 h/c tile
  bool ev1 = (tid + THREADS) < ROWS*HID;          // slot 1 valid (tid < 512)
  int  er0 = tid / HID,            ehc0 = tid - er0*HID;
  int  i1  = tid + THREADS;
  int  er1 = i1 / HID,             ehc1 = i1 - er1*HID;
  float c0 = 0.0f, c1 = 0.0f;                     // cell state lives in registers

  // ---- init LDS: zero A buffers (h0 = 0, K padding = 0) ----
  for (int i = tid; i < 2*ROWS*AST; i += THREADS)
    ((unsigned short*)Abuf)[i] = 0;
  // stage x_0
  Abuf[0][xr0][xk0] = f2bf(xp0[-DIM]);
  if (xv1) Abuf[0][xr1][xk1] = f2bf(xp1[-DIM]);
  __syncthreads();

  // C/D layout: VGPR r -> M = r + 8*(lane>>4), N = lane&15
  const int crow = 8 * (lane >> 4);
  const int ccol = wid * 16 + nl;
  const int arow = lane & 15;
  const int asub = (lane >> 4) * 16;

  for (int t = 0; t < SEQT; ++t) {
    const int cur = t & 1, nxt = cur ^ 1;

    // (a) Issue global loads of x_{t+1} first: latency overlaps the WMMA phase.
    float xreg0 = 0.f, xreg1 = 0.f;
    if (t + 1 < SEQT) {
      xreg0 = *xp0;
      if (xv1) xreg1 = *xp1;
      if (t + 2 < SEQT) {
        __builtin_prefetch(xp0 + DIM, 0, 1);       // global_prefetch_b8
        if (xv1) __builtin_prefetch(xp1 + DIM, 0, 1);
      }
      xp0 += DIM; xp1 += DIM;
    }

    // (b) gates(16x272) = [x_t | h_t](16x160 bf16) @ W^T(160x272) + bias.
    //     Software-pipelined fragment loads: ld(kt+1) issued before wmma(kt).
    Frag16 afA, afB;
    {
      const uint4* ap = (const uint4*)&Abuf[cur][arow][asub];
      afA.q[0] = ap[0];
      afA.q[1] = ap[1];
    }
    v8f acc = {bsum,bsum,bsum,bsum,bsum,bsum,bsum,bsum};
    #pragma unroll
    for (int kt = 0; kt < KT; ++kt) {
      if (kt + 1 < KT) {
        const uint4* ap = (const uint4*)&Abuf[cur][arow][(kt+1)*32 + asub];
        afB.q[0] = ap[0];      // K asub .. asub+7
        afB.q[1] = ap[1];      // K asub+8 .. asub+15  (+16 bytes)
      }
      acc = __builtin_amdgcn_wmma_f32_16x16x32_bf16(
              false, afA.v, false, bf[kt].v, (short)0, acc, false, false);
      afA = afB;
    }

    #pragma unroll
    for (int r = 0; r < 8; ++r)
      gatesL[crow + r][ccol] = acc[r];

    // (c) retire the staged x_{t+1} into the next A buffer (bf16)
    if (t + 1 < SEQT) {
      Abuf[nxt][xr0][xk0] = f2bf(xreg0);
      if (xv1) Abuf[nxt][xr1][xk1] = f2bf(xreg1);
    }
    __syncthreads();

    // (d) LSTM cell update (c in registers); new h -> bf16 into next A buffer
    {
      float ig = fsig (gatesL[er0][ehc0]);
      float fg = fsig (gatesL[er0][HID   + ehc0]);
      float gg = ftanh(gatesL[er0][2*HID + ehc0]);
      float og = fsig (gatesL[er0][3*HID + ehc0]);
      c0 = fg * c0 + ig * gg;
      float hv = og * ftanh(c0);
      Abuf[nxt][er0][DIM + ehc0] = f2bf(hv);
      if (t == SEQT-1) hOut[(size_t)(b0+er0)*HID + ehc0] = hv;
    }
    if (ev1) {
      float ig = fsig (gatesL[er1][ehc1]);
      float fg = fsig (gatesL[er1][HID   + ehc1]);
      float gg = ftanh(gatesL[er1][2*HID + ehc1]);
      float og = fsig (gatesL[er1][3*HID + ehc1]);
      c1 = fg * c1 + ig * gg;
      float hv = og * ftanh(c1);
      Abuf[nxt][er1][DIM + ehc1] = f2bf(hv);
      if (t == SEQT-1) hOut[(size_t)(b0+er1)*HID + ehc1] = hv;
    }
    __syncthreads();
  }
}

// BatchNorm (batch statistics, biased variance) + final Linear(H->1).
__global__ void bn_fc(const float* __restrict__ hT, const float* __restrict__ gamma,
                      const float* __restrict__ beta, const float* __restrict__ fcw,
                      const float* __restrict__ fcb, float* __restrict__ out) {
  __shared__ float meanL[HID], rstdL[HID], gL[HID], bL[HID], wL[HID];
  const int tid = threadIdx.x;
  if (tid < HID) {
    float s = 0.f, ss = 0.f;
    for (int b = 0; b < BATCH; ++b) {
      float v = hT[b*HID + tid];
      s += v; ss += v*v;
    }
    float m   = s * (1.0f/BATCH);
    float var = ss * (1.0f/BATCH) - m*m;
    meanL[tid] = m;
    rstdL[tid] = rsqrtf(var + 1e-5f);
    gL[tid] = gamma[tid]; bL[tid] = beta[tid]; wL[tid] = fcw[tid];
  }
  __syncthreads();
  for (int b = tid; b < BATCH; b += 256) {
    float acc = fcb[0];
    #pragma unroll 6
    for (int hc = 0; hc < HID; ++hc)
      acc += ((hT[b*HID + hc] - meanL[hc]) * rstdL[hc] * gL[hc] + bL[hc]) * wL[hc];
    out[b] = acc;
  }
}

extern "C" void kernel_launch(void* const* d_in, const int* in_sizes, int n_in,
                              void* d_out, int out_size, void* d_ws, size_t ws_size,
                              hipStream_t stream) {
  const float* x     = (const float*)d_in[0];
  const float* Wih   = (const float*)d_in[1];
  const float* Whh   = (const float*)d_in[2];
  const float* bih   = (const float*)d_in[3];
  const float* bhh   = (const float*)d_in[4];
  const float* gamma = (const float*)d_in[5];
  const float* beta  = (const float*)d_in[6];
  const float* fcw   = (const float*)d_in[7];
  const float* fcb   = (const float*)d_in[8];
  float* out = (float*)d_out;                 // [512] logits
  float* hT  = (float*)d_ws;                  // [512*66] f32 final hidden state

  lstm_fused<<<BATCH/ROWS, THREADS, 0, stream>>>(x, Wih, Whh, bih, bhh, hT);
  bn_fc<<<1, 256, 0, stream>>>(hT, gamma, beta, fcw, fcb, out);
}
